// GRU_62234076119437
// MI455X (gfx1250) — matrix operand order repaired and checked by reading, not compiled, not run on hardware
//
#include <hip/hip_runtime.h>
#include <hip/hip_bf16.h>

typedef __bf16 bf16_t;
typedef __attribute__((ext_vector_type(16))) __bf16 v16bf;
typedef __attribute__((ext_vector_type(8)))  __bf16 v8bf;
typedef __attribute__((ext_vector_type(8)))  float  v8f;
typedef __attribute__((ext_vector_type(4)))  float  v4f;

#define T_STEPS  512
#define BATCH    128
#define DIM      1024            // I == Hd == O == 1024
#define MSZ      (1024 * 1024)   // elements per packed weight matrix
#define NBLOCKS  64
#define NTHREADS 256
#define NWAVES   (NBLOCKS * 8)   // wave32 -> 8 waves per 256-thread block

// ---------------------------------------------------------------------------
// Device-wide barrier (persistent kernel, L2 atomics, agent scope)
// ---------------------------------------------------------------------------
__device__ __forceinline__ void grid_barrier(unsigned* cnt, unsigned* gen) {
    __threadfence();            // agent-scope release of this thread's stores
    __syncthreads();
    if (threadIdx.x == 0) {
        unsigned g = __hip_atomic_load(gen, __ATOMIC_RELAXED, __HIP_MEMORY_SCOPE_AGENT);
        unsigned prev = __hip_atomic_fetch_add(cnt, 1u, __ATOMIC_ACQ_REL, __HIP_MEMORY_SCOPE_AGENT);
        if (prev == (unsigned)(NBLOCKS - 1)) {
            __hip_atomic_store(cnt, 0u, __ATOMIC_RELAXED, __HIP_MEMORY_SCOPE_AGENT);
            __hip_atomic_fetch_add(gen, 1u, __ATOMIC_ACQ_REL, __HIP_MEMORY_SCOPE_AGENT);
        } else {
            while (__hip_atomic_load(gen, __ATOMIC_ACQUIRE, __HIP_MEMORY_SCOPE_AGENT) == g) {
                __builtin_amdgcn_s_sleep(2);
            }
        }
    }
    __syncthreads();
}

__device__ __forceinline__ float sigmoidf_(float x) {
    return 1.0f / (1.0f + __expf(-x));
}
__device__ __forceinline__ float tanhf_(float x) {
    x = fminf(15.0f, fmaxf(-15.0f, x));
    float e = __expf(2.0f * x);
    return (e - 1.0f) / (e + 1.0f);
}

// ---------------------------------------------------------------------------
// WMMA fragment helpers.
// B (weights) pre-swizzled: tile (tk,tn) at ((tn*32+tk)*512), lane's 16 bf16
// contiguous.  A: row-major; lanes 0-15 hold K 0-7 / 16-23, lanes 16-31 hold
// K 8-15 / 24-31 (per CDNA5 ISA 16-bit A layout).
// ---------------------------------------------------------------------------
__device__ __forceinline__ v16bf load_b(const bf16_t* Wp, int tk, int tn, int lane) {
    return *(const v16bf*)(Wp + (((size_t)((tn << 5) + tk)) << 9) + (lane << 4));
}

// acc[4] += A[m0:m0+16, :] (bf16 row-major, ld=DIM) * W[:, q*64 : q*64+64]
__device__ __forceinline__ void gemm_a16(v8f acc[4], const bf16_t* A, const bf16_t* Wp,
                                         int m0, int q, int lane) {
    int row  = m0 + (lane & 15);
    int koff = (lane >> 4) << 3;
    const bf16_t* pa = A + (size_t)row * DIM + koff;
#pragma unroll 2
    for (int tk = 0; tk < 32; ++tk) {
        v8bf lo = *(const v8bf*)(pa);
        v8bf hi = *(const v8bf*)(pa + 16);
        v16bf a;
#pragma unroll
        for (int i = 0; i < 8; ++i) { a[i] = lo[i]; a[i + 8] = hi[i]; }
#pragma unroll
        for (int j = 0; j < 4; ++j) {
            v16bf b = load_b(Wp, tk, (q << 2) + j, lane);
            acc[j] = __builtin_amdgcn_wmma_f32_16x16x32_bf16(
                false, a, false, b, (short)0, acc[j], false, false);
        }
        pa += 32;
    }
}

// Same, but A is f32 row-major (the x_t activations), converted on the fly.
__device__ __forceinline__ void gemm_a32(v8f acc[4], const float* A, const bf16_t* Wp,
                                         int m0, int q, int lane) {
    int row  = m0 + (lane & 15);
    int koff = (lane >> 4) << 3;
    const float* pa = A + (size_t)row * DIM + koff;
#pragma unroll 2
    for (int tk = 0; tk < 32; ++tk) {
        v4f x0 = *(const v4f*)(pa);
        v4f x1 = *(const v4f*)(pa + 4);
        v4f x2 = *(const v4f*)(pa + 16);
        v4f x3 = *(const v4f*)(pa + 20);
        v16bf a;
#pragma unroll
        for (int i = 0; i < 4; ++i) {
            a[i]      = (__bf16)x0[i];
            a[4 + i]  = (__bf16)x1[i];
            a[8 + i]  = (__bf16)x2[i];
            a[12 + i] = (__bf16)x3[i];
        }
#pragma unroll
        for (int j = 0; j < 4; ++j) {
            v16bf b = load_b(Wp, tk, (q << 2) + j, lane);
            acc[j] = __builtin_amdgcn_wmma_f32_16x16x32_bf16(
                false, a, false, b, (short)0, acc[j], false, false);
        }
        pa += 32;
    }
}

// ---------------------------------------------------------------------------
// Weight packer: f32 [K=1024][N=1024] row-major -> swizzled bf16 WMMA tiles.
// Packed element p of a matrix: tile=p>>9, lane=(p>>4)&31, e=p&15;
// n = tn*16 + (lane&15); k = tk*32 + (lane>=16 ? 8 : 0) + (e&7) + (e>=8 ? 16 : 0).
// Also (re)initializes the grid-barrier state each call.
// ---------------------------------------------------------------------------
__global__ void gru_pack_weights(const float* w0, const float* w1, const float* w2,
                                 const float* w3, const float* w4, const float* w5,
                                 const float* w6, bf16_t* dst,
                                 unsigned* cnt, unsigned* gen) {
    size_t idx = (size_t)blockIdx.x * blockDim.x + threadIdx.x;
    if (idx == 0) { *cnt = 0u; *gen = 0u; }
    if (idx >= (size_t)7 * MSZ) return;
    int mat = (int)(idx / MSZ);
    int p   = (int)(idx % MSZ);
    int tile = p >> 9;
    int l    = (p >> 4) & 31;
    int e    = p & 15;
    int tn = tile >> 5;
    int tk = tile & 31;
    int n = (tn << 4) + (l & 15);
    int k = (tk << 5) + ((l >> 4) << 3) + (e & 7) + ((e >> 3) << 4);
    const float* W = (mat == 0) ? w0 : (mat == 1) ? w1 : (mat == 2) ? w2
                   : (mat == 3) ? w3 : (mat == 4) ? w4 : (mat == 5) ? w5 : w6;
    dst[idx] = (bf16_t)W[(size_t)k * 1024 + n];
}

// ---------------------------------------------------------------------------
// Persistent GRU scan.  Per step t:
//   phase 1: Z = sig(x@Wxz + h@Whz + bz); R = sig(x@Wxr + h@Whr + br);
//            Rh = R*h (bf16);  y_{t-1} = h@Whq + bq  (deferred output GEMM)
//   phase 2: Ht = tanh(x@Wxh + Rh@Whh + bh); h = Z*h + (1-Z)*Ht
// Epilogue: y_{T-1}, H_final.
// ---------------------------------------------------------------------------
__global__ __launch_bounds__(NTHREADS, 1)
void gru_scan(const float* __restrict__ x, const float* __restrict__ H0,
              const float* __restrict__ bz, const float* __restrict__ br,
              const float* __restrict__ bh, const float* __restrict__ bq,
              const bf16_t* __restrict__ Wp,
              float* h32, bf16_t* h16, bf16_t* rh16, float* z32,
              float* __restrict__ out, unsigned* cnt, unsigned* gen) {
    const bf16_t* Wxz = Wp + 0 * (size_t)MSZ;
    const bf16_t* Whz = Wp + 1 * (size_t)MSZ;
    const bf16_t* Wxr = Wp + 2 * (size_t)MSZ;
    const bf16_t* Whr = Wp + 3 * (size_t)MSZ;
    const bf16_t* Wxh = Wp + 4 * (size_t)MSZ;
    const bf16_t* Whh = Wp + 5 * (size_t)MSZ;
    const bf16_t* Whq = Wp + 6 * (size_t)MSZ;

    const int wv   = (blockIdx.x << 3) + (threadIdx.x >> 5);
    const int lane = threadIdx.x & 31;
    const int tid  = blockIdx.x * NTHREADS + threadIdx.x;

    // init hidden state (f32 master + bf16 shadow)
    for (int i = tid; i < BATCH * DIM; i += NBLOCKS * NTHREADS) {
        float v = H0[i];
        h32[i] = v;
        h16[i] = (bf16_t)v;
    }
    grid_barrier(cnt, gen);

    for (int t = 0; t < T_STEPS; ++t) {
        const float* xt = x + (size_t)t * BATCH * DIM;

        // ---------------- phase 1: Z, R, (deferred) y_{t-1} ----------------
        for (int task = wv; task < 384; task += NWAVES) {
            int kind = task >> 7;     // 0=Z, 1=R, 2=Y
            int id   = task & 127;
            int m0 = (id >> 4) << 4;  // 8 row tiles
            int q  = id & 15;         // 16 column quads (64 cols each)
            if (kind == 2 && t == 0) continue;
            v8f acc[4] = {};
            if (kind == 0) {
                gemm_a32(acc, xt, Wxz, m0, q, lane);
                gemm_a16(acc, h16, Whz, m0, q, lane);
            } else if (kind == 1) {
                gemm_a32(acc, xt, Wxr, m0, q, lane);
                gemm_a16(acc, h16, Whr, m0, q, lane);
            } else {
                gemm_a16(acc, h16, Whq, m0, q, lane);
            }
            int mb = m0 + ((lane >> 4) << 3);   // D layout: rows v + 8*(lane>=16)
#pragma unroll
            for (int j = 0; j < 4; ++j) {
                int c = (q << 6) + (j << 4) + (lane & 15);
                if (kind == 0) {
                    float bias = bz[c];
#pragma unroll
                    for (int v = 0; v < 8; ++v) {
                        int m = mb + v;
                        z32[m * DIM + c] = sigmoidf_(acc[j][v] + bias);
                    }
                } else if (kind == 1) {
                    float bias = br[c];
#pragma unroll
                    for (int v = 0; v < 8; ++v) {
                        int m = mb + v;
                        float r = sigmoidf_(acc[j][v] + bias);
                        rh16[m * DIM + c] = (bf16_t)(r * h32[m * DIM + c]);
                    }
                } else {
                    float bias = bq[c];
#pragma unroll
                    for (int v = 0; v < 8; ++v) {
                        int m = mb + v;
                        out[((size_t)(t - 1) * BATCH + m) * DIM + c] = acc[j][v] + bias;
                    }
                }
            }
        }
        grid_barrier(cnt, gen);

        // ---------------- phase 2: H~ and hidden-state update --------------
        for (int task = wv; task < 128; task += NWAVES) {
            int m0 = (task >> 4) << 4;
            int q  = task & 15;
            v8f acc[4] = {};
            gemm_a32(acc, xt, Wxh, m0, q, lane);
            gemm_a16(acc, rh16, Whh, m0, q, lane);
            int mb = m0 + ((lane >> 4) << 3);
#pragma unroll
            for (int j = 0; j < 4; ++j) {
                int c = (q << 6) + (j << 4) + (lane & 15);
                float bias = bh[c];
#pragma unroll
                for (int v = 0; v < 8; ++v) {
                    int m = mb + v;
                    float ht = tanhf_(acc[j][v] + bias);
                    float zv = z32[m * DIM + c];
                    float hv = h32[m * DIM + c];
                    float hn = zv * hv + (1.0f - zv) * ht;
                    h32[m * DIM + c] = hn;
                    h16[m * DIM + c] = (bf16_t)hn;
                }
            }
        }
        grid_barrier(cnt, gen);
    }

    // ---------------- epilogue: y_{T-1} and H_final ------------------------
    for (int task = wv; task < 128; task += NWAVES) {
        int m0 = (task >> 4) << 4;
        int q  = task & 15;
        v8f acc[4] = {};
        gemm_a16(acc, h16, Whq, m0, q, lane);
        int mb = m0 + ((lane >> 4) << 3);
#pragma unroll
        for (int j = 0; j < 4; ++j) {
            int c = (q << 6) + (j << 4) + (lane & 15);
            float bias = bq[c];
#pragma unroll
            for (int v = 0; v < 8; ++v) {
                int m = mb + v;
                out[((size_t)(T_STEPS - 1) * BATCH + m) * DIM + c] = acc[j][v] + bias;
            }
        }
    }
    for (int i = tid; i < BATCH * DIM; i += NBLOCKS * NTHREADS) {
        out[(size_t)T_STEPS * BATCH * DIM + i] = h32[i];
    }
}

// ---------------------------------------------------------------------------
// Host launcher.  Input order: inputs, H, W_xz, W_hz, b_z, W_xr, W_hr, b_r,
// W_xh, W_hh, b_h, W_hq, b_q.  Output: Ys [T,B,O] then H_final [B,Hd].
// ---------------------------------------------------------------------------
extern "C" void kernel_launch(void* const* d_in, const int* in_sizes, int n_in,
                              void* d_out, int out_size, void* d_ws, size_t ws_size,
                              hipStream_t stream) {
    const float* x   = (const float*)d_in[0];
    const float* H0  = (const float*)d_in[1];
    const float* Wxz = (const float*)d_in[2];
    const float* Whz = (const float*)d_in[3];
    const float* bz  = (const float*)d_in[4];
    const float* Wxr = (const float*)d_in[5];
    const float* Whr = (const float*)d_in[6];
    const float* br  = (const float*)d_in[7];
    const float* Wxh = (const float*)d_in[8];
    const float* Whh = (const float*)d_in[9];
    const float* bh  = (const float*)d_in[10];
    const float* Whq = (const float*)d_in[11];
    const float* bq  = (const float*)d_in[12];

    char* ws = (char*)d_ws;
    const size_t OFF_WP  = 0;
    const size_t OFF_H32 = OFF_WP  + (size_t)7 * MSZ * 2;        // 14,680,064
    const size_t OFF_H16 = OFF_H32 + (size_t)BATCH * DIM * 4;
    const size_t OFF_RH  = OFF_H16 + (size_t)BATCH * DIM * 2;
    const size_t OFF_Z32 = OFF_RH  + (size_t)BATCH * DIM * 2;
    const size_t OFF_BAR = OFF_Z32 + (size_t)BATCH * DIM * 4;

    bf16_t*   Wp   = (bf16_t*)(ws + OFF_WP);
    float*    h32  = (float*)(ws + OFF_H32);
    bf16_t*   h16  = (bf16_t*)(ws + OFF_H16);
    bf16_t*   rh16 = (bf16_t*)(ws + OFF_RH);
    float*    z32  = (float*)(ws + OFF_Z32);
    unsigned* cnt  = (unsigned*)(ws + OFF_BAR);
    unsigned* gen  = cnt + 1;

    const int pack_threads = 256;
    const int pack_blocks  = (7 * MSZ + pack_threads - 1) / pack_threads;
    gru_pack_weights<<<pack_blocks, pack_threads, 0, stream>>>(
        Wxz, Whz, Wxr, Whr, Wxh, Whh, Whq, Wp, cnt, gen);

    gru_scan<<<NBLOCKS, NTHREADS, 0, stream>>>(
        x, H0, bz, br, bh, bq, Wp, h32, h16, rh16, z32,
        (float*)d_out, cnt, gen);
}
